// CombinedGraphModel_84670985274089
// MI455X (gfx1250) — matrix operand order repaired and checked by reading, not compiled, not run on hardware
//
#include <hip/hip_runtime.h>
#include <hip/hip_bf16.h>

// ---------------------------------------------------------------------------
// CDNA5 (gfx1250) GNN pipeline: SAGE -> GAT(4 heads) -> SAGE -> GAT(1 head) -> dense
// GEMMs use v_wmma_f32_16x16x32_bf16 (one wave per 16x16 tile, f32 accumulate).
// Weights are pre-packed per launch into bf16 B-fragment order so each lane
// loads its whole fragment with two global_load_b128. A fragments are
// converted f32->bf16 with the native v_cvt hardware path.
// Graph aggregation uses global_atomic_add_f32 / atomic_max_u32 (L2-resident).
// ---------------------------------------------------------------------------

typedef __attribute__((ext_vector_type(16))) __bf16 v16bf;
typedef __attribute__((ext_vector_type(8)))  float  v8f;

// native hardware convert (v_cvt_bf16_f32, round-to-nearest-even)
__device__ __forceinline__ __bf16 f2bf(float f) { return static_cast<__bf16>(f); }

__device__ __forceinline__ float lrelu(float x) { return x > 0.f ? x : 0.2f * x; }

// order-preserving float <-> uint encoding (for atomicMax-based segment max)
__device__ __forceinline__ unsigned fenc(float f) {
  unsigned u = __float_as_uint(f);
  return (u & 0x80000000u) ? ~u : (u | 0x80000000u);
}
__device__ __forceinline__ float fdec(unsigned k) {
  unsigned u = (k & 0x80000000u) ? (k & 0x7FFFFFFFu) : ~k;
  return __uint_as_float(u);
}

// ---------------------------------------------------------------------------
// Pack W[K x NC] f32 row-major into bf16 WMMA-B-fragment order:
//   Wp[(kg*NC + c)*2 + half][i]  =  bf16( W[kg*32 + 8*half + (i&7) + 16*(i>>3)][c] )
// so a lane's 16-element fragment (K-half selected by lane>>4) is 32
// contiguous bytes.
// ---------------------------------------------------------------------------
__global__ void k_pack_w(const float* __restrict__ W, __bf16* __restrict__ Wp,
                         int K, int NC) {
  int t = blockIdx.x * blockDim.x + threadIdx.x;
  if (t >= K * NC) return;
  int k = t / NC, c = t - k * NC;
  int kg = k >> 5, kr = k & 31;
  int half = (kr >> 3) & 1, i = (kr & 7) + ((kr >> 4) << 3);
  Wp[(((size_t)kg * NC + c) * 2 + half) * 16 + i] = f2bf(W[t]);
}

// ---------------------------------------------------------------------------
// WMMA GEMM:  out[M x NC] = act( A1 @ W1 (+ A2 @ W2) + bias )
// A: [M x K] f32 row-major (converted to bf16 on the fly),
// W*: packed bf16 fragments (see k_pack_w).
// One wave -> one 16x16 tile. Block = 128 threads = 4 waves = 64 columns.
// Requires M%16==0, K%32==0, NC%64==0 (all exact here: M=50000).
// ---------------------------------------------------------------------------
__global__ void __launch_bounds__(128)
k_gemm_wmma(const float* __restrict__ A1, const __bf16* __restrict__ W1p,
            const float* __restrict__ A2, const __bf16* __restrict__ W2p,
            const float* __restrict__ bias, float* __restrict__ out,
            int K, int NC, int relu_flag)
{
  const int lane = threadIdx.x & 31;
  const int wave = threadIdx.x >> 5;
  const int row0 = blockIdx.x << 4;
  const int col0 = (blockIdx.y << 6) + (wave << 4);
  const int r    = row0 + (lane & 15);
  const int c    = col0 + (lane & 15);
  const int kA   = (lane >> 4) << 3;   // 0 or 8 (A fragment K sub-offset)
  const int half = lane >> 4;          // B fragment K half

  v8f acc = {0.f, 0.f, 0.f, 0.f, 0.f, 0.f, 0.f, 0.f};

  for (int k0 = 0; k0 < K; k0 += 32) {
    const size_t boff = (((size_t)(k0 >> 5) * NC + c) * 2 + half) * 16;
    // ---- A1/W1 fragment ----
    {
      const float* Ar = A1 + (size_t)r * K + k0;
      float4 a0 = *(const float4*)(Ar + kA);
      float4 a1 = *(const float4*)(Ar + kA + 4);
      float4 a2 = *(const float4*)(Ar + 16 + kA);
      float4 a3 = *(const float4*)(Ar + 16 + kA + 4);
      v16bf a;
      a[0]=f2bf(a0.x); a[1]=f2bf(a0.y); a[2]=f2bf(a0.z); a[3]=f2bf(a0.w);
      a[4]=f2bf(a1.x); a[5]=f2bf(a1.y); a[6]=f2bf(a1.z); a[7]=f2bf(a1.w);
      a[8]=f2bf(a2.x); a[9]=f2bf(a2.y); a[10]=f2bf(a2.z); a[11]=f2bf(a2.w);
      a[12]=f2bf(a3.x); a[13]=f2bf(a3.y); a[14]=f2bf(a3.z); a[15]=f2bf(a3.w);
      v16bf b = *(const v16bf*)(W1p + boff);   // 32 contiguous bytes
      acc = __builtin_amdgcn_wmma_f32_16x16x32_bf16(false, a, false, b,
                                                    (short)0, acc, false, false);
    }
    // ---- optional fused A2/W2 fragment (SAGE: mean@Wl + x@Wr) ----
    if (A2) {
      const float* Ar = A2 + (size_t)r * K + k0;
      float4 a0 = *(const float4*)(Ar + kA);
      float4 a1 = *(const float4*)(Ar + kA + 4);
      float4 a2 = *(const float4*)(Ar + 16 + kA);
      float4 a3 = *(const float4*)(Ar + 16 + kA + 4);
      v16bf a;
      a[0]=f2bf(a0.x); a[1]=f2bf(a0.y); a[2]=f2bf(a0.z); a[3]=f2bf(a0.w);
      a[4]=f2bf(a1.x); a[5]=f2bf(a1.y); a[6]=f2bf(a1.z); a[7]=f2bf(a1.w);
      a[8]=f2bf(a2.x); a[9]=f2bf(a2.y); a[10]=f2bf(a2.z); a[11]=f2bf(a2.w);
      a[12]=f2bf(a3.x); a[13]=f2bf(a3.y); a[14]=f2bf(a3.z); a[15]=f2bf(a3.w);
      v16bf b = *(const v16bf*)(W2p + boff);
      acc = __builtin_amdgcn_wmma_f32_16x16x32_bf16(false, a, false, b,
                                                    (short)0, acc, false, false);
    }
  }

  // C/D layout: VGPR j -> row = row0 + 8*(lane>>4) + j, col = c
  const int   orow = row0 + (half << 3);
  const float bv   = bias ? bias[c] : 0.f;
#pragma unroll
  for (int j = 0; j < 8; ++j) {
    float v = acc[j] + bv;
    if (relu_flag) v = fmaxf(v, 0.f);
    out[(size_t)(orow + j) * NC + c] = v;
  }
}

// ---------------------------------------------------------------------------
// Graph kernels
// ---------------------------------------------------------------------------
__global__ void k_deg(const int* __restrict__ dst, float* __restrict__ deg, int E) {
  int e = blockIdx.x * blockDim.x + threadIdx.x;
  if (e < E) atomicAdd(&deg[dst[e]], 1.0f);
}

// block per edge, thread per feature (blockDim == F)
__global__ void k_scatter_add(const float* __restrict__ xin, const int* __restrict__ src,
                              const int* __restrict__ dst, float* __restrict__ agg) {
  int e = blockIdx.x, f = threadIdx.x, F = blockDim.x;
  int s = src[e], d = dst[e];
  atomicAdd(&agg[(size_t)d * F + f], xin[(size_t)s * F + f]);
}

// mean = agg / max(deg,1), in place (block per node, thread per feature)
__global__ void k_scale_mean(float* __restrict__ agg, const float* __restrict__ deg) {
  int n = blockIdx.x, f = threadIdx.x, F = blockDim.x;
  float inv = 1.0f / fmaxf(deg[n], 1.0f);
  agg[(size_t)n * F + f] *= inv;
}

// per-node attention logit halves: alpha_s[n,h] = <xp[n,h,:], a_src[h,:]>
__global__ void k_alpha(const float* __restrict__ xp, const float* __restrict__ asr,
                        const float* __restrict__ adt, float* __restrict__ alS,
                        float* __restrict__ alD, int N, int H, int C) {
  int t = blockIdx.x * blockDim.x + threadIdx.x;
  if (t >= N * H) return;
  int n = t / H, h = t - n * H;
  const float* row = xp + (size_t)n * H * C + (size_t)h * C;
  const float* ws_ = asr + (size_t)h * C;
  const float* wd_ = adt + (size_t)h * C;
  float ss = 0.f, sd = 0.f;
  for (int cc = 0; cc < C; ++cc) { float v = row[cc]; ss += v * ws_[cc]; sd += v * wd_[cc]; }
  alS[t] = ss;
  alD[t] = sd;
}

// pass 1: segment max of leaky_relu(alpha_s[s]+alpha_d[d]) over incoming edges
__global__ void k_gat_max(const int* __restrict__ src, const int* __restrict__ dst,
                          const float* __restrict__ alS, const float* __restrict__ alD,
                          unsigned* __restrict__ emax, int E, int H, int total) {
  int t = blockIdx.x * blockDim.x + threadIdx.x;
  if (t >= total) return;
  int e = t / H, h = t - e * H;
  int s, d;
  if (e < E) { s = src[e]; d = dst[e]; } else { s = d = e - E; }  // self loops
  float ev = lrelu(alS[s * H + h] + alD[d * H + h]);
  atomicMax(&emax[d * H + h], fenc(ev));
}

// pass 2: softmax denominator
__global__ void k_gat_denom(const int* __restrict__ src, const int* __restrict__ dst,
                            const float* __restrict__ alS, const float* __restrict__ alD,
                            const unsigned* __restrict__ emax, float* __restrict__ denom,
                            int E, int H, int total) {
  int t = blockIdx.x * blockDim.x + threadIdx.x;
  if (t >= total) return;
  int e = t / H, h = t - e * H;
  int s, d;
  if (e < E) { s = src[e]; d = dst[e]; } else { s = d = e - E; }
  float ev = lrelu(alS[s * H + h] + alD[d * H + h]);
  atomicAdd(&denom[d * H + h], expf(ev - fdec(emax[d * H + h])));
}

// pass 3: weighted scatter  acc[d,f] += softmax_w * xp[s,f]   (block/edge, thread/feature)
__global__ void k_gat_agg(const int* __restrict__ src, const int* __restrict__ dst,
                          const float* __restrict__ alS, const float* __restrict__ alD,
                          const unsigned* __restrict__ emax, const float* __restrict__ denom,
                          const float* __restrict__ xp, float* __restrict__ acc,
                          int E, int H, int C) {
  int e = blockIdx.x, f = threadIdx.x;
  int h = f / C;
  int s, d;
  if (e < E) { s = src[e]; d = dst[e]; } else { s = d = e - E; }
  float ev = lrelu(alS[s * H + h] + alD[d * H + h]);
  float w  = expf(ev - fdec(emax[d * H + h])) / denom[d * H + h];
  int HC = H * C;
  atomicAdd(&acc[(size_t)d * HC + f], w * xp[(size_t)s * HC + f]);
}

__global__ void k_bias_act(const float* __restrict__ acc, const float* __restrict__ b,
                           float* __restrict__ outp, int total, int F, int relu_flag) {
  int t = blockIdx.x * blockDim.x + threadIdx.x;
  if (t >= total) return;
  float v = acc[t] + b[t % F];
  if (relu_flag) v = fmaxf(v, 0.f);
  outp[t] = v;
}

// ---------------------------------------------------------------------------
extern "C" void kernel_launch(void* const* d_in, const int* in_sizes, int n_in,
                              void* d_out, int out_size, void* d_ws, size_t ws_size,
                              hipStream_t stream) {
  (void)n_in; (void)out_size; (void)ws_size;
  const int IN = 128, HID = 64, HEADS = 4, OUT = 64;

  const float* x   = (const float*)d_in[0];
  const int*   ei  = (const int*)d_in[1];
  const int    N   = in_sizes[0] / IN;   // 50000 (multiple of 16)
  const int    E   = in_sizes[1] / 2;    // 800000
  const int*   src = ei;
  const int*   dst = ei + E;

  const float *s1Wl = (const float*)d_in[2],  *s1b = (const float*)d_in[3],
              *s1Wr = (const float*)d_in[4];
  const float *g1W  = (const float*)d_in[5],  *g1as = (const float*)d_in[6],
              *g1ad = (const float*)d_in[7],  *g1b  = (const float*)d_in[8];
  const float *s2Wl = (const float*)d_in[9],  *s2b  = (const float*)d_in[10],
              *s2Wr = (const float*)d_in[11];
  const float *g2W  = (const float*)d_in[12], *g2as = (const float*)d_in[13],
              *g2ad = (const float*)d_in[14], *g2b  = (const float*)d_in[15];
  const float *dW   = (const float*)d_in[16], *db   = (const float*)d_in[17];

  // output tuple layout: (h_dense, h_sage1, h_gat1, h_sage2)
  float* out_dense = (float*)d_out;
  float* h_sage1   = out_dense + (size_t)N * OUT;
  float* h_gat1    = h_sage1 + (size_t)N * HID;
  float* h_sage2   = h_gat1 + (size_t)N * HEADS * HID;

  // workspace layout (~157 MB + 144 KB packed weights), buffers reused
  float*    ws    = (float*)d_ws;
  float*    deg   = ws;                          // N
  float*    agg   = deg + N;                     // N*256 (also holds h_gat2)
  float*    xp    = agg + (size_t)N * 256;       // N*256
  float*    acc   = xp + (size_t)N * 256;        // N*256
  float*    alS   = acc + (size_t)N * 256;       // N*HEADS
  float*    alD   = alS + (size_t)N * HEADS;     // N*HEADS
  unsigned* emax  = (unsigned*)(alD + (size_t)N * HEADS);  // N*HEADS
  float*    denom = (float*)(emax + (size_t)N * HEADS);    // N*HEADS

  // packed bf16 weight fragments (each size multiple of 16 elems -> 32B aligned)
  __bf16* wp = (__bf16*)(denom + (size_t)N * HEADS);
  __bf16* p_s1Wl = wp; wp += 128 * 64;
  __bf16* p_s1Wr = wp; wp += 128 * 64;
  __bf16* p_g1W  = wp; wp += 64 * 256;
  __bf16* p_s2Wl = wp; wp += 256 * 64;
  __bf16* p_s2Wr = wp; wp += 256 * 64;
  __bf16* p_g2W  = wp; wp += 64 * 64;
  __bf16* p_dW   = wp; wp += 64 * 64;

  const int Etot = E + N;        // edges + self loops
  const int HC1  = HEADS * HID;  // 256

  // ---------------- pack all weights (tiny, once per launch) ----------------
  k_pack_w<<<(128 * 64 + 255) / 256, 256, 0, stream>>>(s1Wl, p_s1Wl, 128, 64);
  k_pack_w<<<(128 * 64 + 255) / 256, 256, 0, stream>>>(s1Wr, p_s1Wr, 128, 64);
  k_pack_w<<<(64 * 256 + 255) / 256, 256, 0, stream>>>(g1W, p_g1W, 64, 256);
  k_pack_w<<<(256 * 64 + 255) / 256, 256, 0, stream>>>(s2Wl, p_s2Wl, 256, 64);
  k_pack_w<<<(256 * 64 + 255) / 256, 256, 0, stream>>>(s2Wr, p_s2Wr, 256, 64);
  k_pack_w<<<(64 * 64 + 255) / 256, 256, 0, stream>>>(g2W, p_g2W, 64, 64);
  k_pack_w<<<(64 * 64 + 255) / 256, 256, 0, stream>>>(dW, p_dW, 64, 64);

  // ---------------- layer 1: SAGE(IN=128 -> HID=64) ----------------
  hipMemsetAsync(deg, 0, (size_t)N * 4, stream);
  hipMemsetAsync(agg, 0, (size_t)N * IN * 4, stream);
  k_deg<<<(E + 255) / 256, 256, 0, stream>>>(dst, deg, E);
  k_scatter_add<<<E, IN, 0, stream>>>(x, src, dst, agg);
  k_scale_mean<<<N, IN, 0, stream>>>(agg, deg);
  k_gemm_wmma<<<dim3(N / 16, HID / 64), 128, 0, stream>>>(
      agg, p_s1Wl, x, p_s1Wr, s1b, h_sage1, IN, HID, 1);

  // ---------------- layer 2: GAT(HID -> HEADS x HID, concat) ----------------
  k_gemm_wmma<<<dim3(N / 16, HC1 / 64), 128, 0, stream>>>(
      h_sage1, p_g1W, nullptr, nullptr, nullptr, xp, HID, HC1, 0);
  k_alpha<<<(N * HEADS + 255) / 256, 256, 0, stream>>>(xp, g1as, g1ad, alS, alD, N, HEADS, HID);
  hipMemsetAsync(emax, 0, (size_t)N * HEADS * 4, stream);
  hipMemsetAsync(denom, 0, (size_t)N * HEADS * 4, stream);
  hipMemsetAsync(acc, 0, (size_t)N * HC1 * 4, stream);
  {
    int tot = Etot * HEADS;
    k_gat_max<<<(tot + 255) / 256, 256, 0, stream>>>(src, dst, alS, alD, emax, E, HEADS, tot);
    k_gat_denom<<<(tot + 255) / 256, 256, 0, stream>>>(src, dst, alS, alD, emax, denom, E, HEADS, tot);
  }
  k_gat_agg<<<Etot, HC1, 0, stream>>>(src, dst, alS, alD, emax, denom, xp, acc, E, HEADS, HID);
  k_bias_act<<<(N * HC1 + 255) / 256, 256, 0, stream>>>(acc, g1b, h_gat1, N * HC1, HC1, 1);

  // ---------------- layer 3: SAGE(256 -> HID=64) ----------------
  hipMemsetAsync(agg, 0, (size_t)N * HC1 * 4, stream);
  k_scatter_add<<<E, HC1, 0, stream>>>(h_gat1, src, dst, agg);
  k_scale_mean<<<N, HC1, 0, stream>>>(agg, deg);
  k_gemm_wmma<<<dim3(N / 16, HID / 64), 128, 0, stream>>>(
      agg, p_s2Wl, h_gat1, p_s2Wr, s2b, h_sage2, HC1, HID, 1);

  // ---------------- layer 4: GAT(HID -> OUT, 1 head, mean) ----------------
  k_gemm_wmma<<<dim3(N / 16, OUT / 64), 128, 0, stream>>>(
      h_sage2, p_g2W, nullptr, nullptr, nullptr, xp, HID, OUT, 0);
  k_alpha<<<(N + 255) / 256, 256, 0, stream>>>(xp, g2as, g2ad, alS, alD, N, 1, OUT);
  hipMemsetAsync(emax, 0, (size_t)N * 4, stream);
  hipMemsetAsync(denom, 0, (size_t)N * 4, stream);
  hipMemsetAsync(acc, 0, (size_t)N * OUT * 4, stream);
  k_gat_max<<<(Etot + 255) / 256, 256, 0, stream>>>(src, dst, alS, alD, emax, E, 1, Etot);
  k_gat_denom<<<(Etot + 255) / 256, 256, 0, stream>>>(src, dst, alS, alD, emax, denom, E, 1, Etot);
  k_gat_agg<<<Etot, OUT, 0, stream>>>(src, dst, alS, alD, emax, denom, xp, acc, E, 1, OUT);
  k_bias_act<<<(N * OUT + 255) / 256, 256, 0, stream>>>(acc, g2b, agg /*h_gat2*/, N * OUT, OUT, 0);

  // ---------------- dense: h_gat2 @ dense_W + dense_b ----------------
  k_gemm_wmma<<<dim3(N / 16, OUT / 64), 128, 0, stream>>>(
      agg, p_dW, nullptr, nullptr, db, out_dense, OUT, OUT, 0);
}
